// IntegralAttention_12249246728989
// MI455X (gfx1250) — matrix operand order repaired and checked by reading, not compiled, not run on hardware
//
#include <hip/hip_runtime.h>

// ---------------------------------------------------------------------------
// Problem constants (from reference): B=2, N=M=2048, C=1024, H=16, HD=64, S=8
// ---------------------------------------------------------------------------
constexpr int Bc  = 2;
constexpr int Nc  = 2048;
constexpr int Mc  = 2048;
constexpr int Cc  = 1024;
constexpr int Hc  = 16;
constexpr int HDc = 64;
// scale = SD^-0.5 / S = (1/sqrt(8))/8
constexpr float kScale = 0.04419417382415922f;

typedef __attribute__((ext_vector_type(16))) __bf16 v16bf;
typedef __attribute__((ext_vector_type(8)))  __bf16 bf16x8;
typedef __attribute__((ext_vector_type(8)))  float  v8f;
typedef __attribute__((ext_vector_type(8)))  float  f32x8;
typedef __attribute__((ext_vector_type(16))) float  f32x16;

__device__ __forceinline__ v8f wmma_bf16(v16bf a, v16bf b, v8f c) {
  // v_wmma_f32_16x16x32_bf16  (8-arg form, probe-confirmed)
  return __builtin_amdgcn_wmma_f32_16x16x32_bf16(
      /*neg_a=*/false, a, /*neg_b=*/false, b,
      /*c_mod=*/(short)0, c, /*reuse_a=*/false, /*reuse_b=*/false);
}

// ---------------------------------------------------------------------------
// Projection GEMM: Y[4096, C] = X[4096, C] @ W^T[C, C] + bias
// MODE 0: X f32 -> Y bf16, row-major [rows, C]                (Q, K)
// MODE 1: X f32 -> Y bf16, per-head transposed [B, H, HD, M]  (V)
// MODE 2: X bf16 -> Y f32, row-major [rows, C]                (output proj)
// One 16x16 output tile per wave, 8 waves per block. X and W tiles are
// re-read across tiles but stay resident in the 192 MB L2.
// ---------------------------------------------------------------------------
template <int MODE>
__global__ void proj_kernel(const float* __restrict__ Xf,
                            const __bf16* __restrict__ Xb,
                            const float* __restrict__ W,
                            const float* __restrict__ bias,
                            __bf16* __restrict__ Yb,
                            float* __restrict__ Yf) {
  const int lane = threadIdx.x & 31;
  const int wave = threadIdx.x >> 5;
  const int tile = blockIdx.x * 8 + wave;
  constexpr int CT = Cc / 16;               // 64 column tiles
  const int m0 = (tile / CT) * 16;          // token-row base
  const int o0 = (tile % CT) * 16;          // out-channel base
  const int hi = lane >> 4;
  const int lo = lane & 15;

  v8f acc = {};
  const float* wrow = W + (size_t)(o0 + lo) * Cc;   // B column = out channel

#pragma unroll 2
  for (int k0 = 0; k0 < Cc; k0 += 32) {
    v16bf a, bb;
    if (MODE == 2) {
      const __bf16* xr = Xb + (size_t)(m0 + lo) * Cc + k0 + hi * 8;
      bf16x8 x0 = *(const bf16x8*)xr;
      bf16x8 x1 = *(const bf16x8*)(xr + 16);
#pragma unroll
      for (int j = 0; j < 8; ++j) { a[j] = x0[j]; a[j + 8] = x1[j]; }
    } else {
      const float* xr = Xf + (size_t)(m0 + lo) * Cc + k0 + hi * 8;
      f32x8 x0 = *(const f32x8*)xr;
      f32x8 x1 = *(const f32x8*)(xr + 16);
#pragma unroll
      for (int j = 0; j < 8; ++j) {
        a[j]     = (__bf16)x0[j];
        a[j + 8] = (__bf16)x1[j];
      }
    }
    f32x8 w0 = *(const f32x8*)(wrow + k0 + hi * 8);
    f32x8 w1 = *(const f32x8*)(wrow + k0 + 16 + hi * 8);
#pragma unroll
    for (int j = 0; j < 8; ++j) { bb[j] = (__bf16)w0[j]; bb[j + 8] = (__bf16)w1[j]; }
    acc = wmma_bf16(a, bb, acc);
  }

  const float bvv = bias[o0 + lo];
  const int col = o0 + lo;
#pragma unroll
  for (int r = 0; r < 8; ++r) {
    const int row = m0 + r + hi * 8;
    const float y = acc[r] + bvv;
    if (MODE == 0) {
      Yb[(size_t)row * Cc + col] = (__bf16)y;
    } else if (MODE == 1) {
      const int b_ = row >> 11;              // row / M
      const int m_ = row & (Mc - 1);
      const int h_ = col >> 6;               // col / HD
      const int d_ = col & (HDc - 1);
      Yb[((size_t)((b_ * Hc + h_) * HDc + d_)) * Mc + m_] = (__bf16)y;
    } else {
      // Final 16 MB f32 output: pure stream-out, keep it out of L2 (NT hint).
      __builtin_nontemporal_store(y, &Yf[(size_t)row * Cc + col]);
    }
  }
}

// ---------------------------------------------------------------------------
// Fused attention: one block per (b, h, 16-query-row tile).
// 16 x 2048 f32 logits live in LDS (128 KB of the 320 KB WGP LDS).
//   Phase 1: QK^T via WMMA (8 waves x 16 cols = 128 cols/iter)
//   Phase 2: row softmax in LDS; write normalized probs to d_out (only pass,
//            non-temporal: 537 MB stream must not evict L2-resident k/v)
//   Phase 3: P.V via WMMA, probs re-read from LDS with 1/sum folded into A
// ---------------------------------------------------------------------------
__global__ void attention_kernel(const __bf16* __restrict__ q_ws,
                                 const __bf16* __restrict__ k_ws,
                                 const __bf16* __restrict__ vt_ws,
                                 __bf16* __restrict__ attn_ws,
                                 float* __restrict__ out) {
  extern __shared__ float smem[];
  float* s_logits = smem;                     // 16*2048
  float* s_red    = smem + 16 * Mc;           // 256
  float* s_stat   = s_red + 256;              // 16 (row max)
  float* s_inv    = s_stat + 16;              // 16 (1/row sum)
  float* s_pv     = s_inv + 16;               // 4*16*16 partial P.V

  constexpr int NT = Nc / 16;                 // 128 row tiles
  const int bt = blockIdx.x;
  const int n0 = (bt % NT) * 16;
  const int h  = (bt / NT) % Hc;
  const int b  = bt / (NT * Hc);

  const int tid  = threadIdx.x;
  const int lane = tid & 31;
  const int wave = tid >> 5;
  const int hi   = lane >> 4;
  const int lo   = lane & 15;

  // ---- Q fragments (A matrix, 16 rows x 64 k => two k-steps), reused 16x --
  const __bf16* qrow = q_ws + (size_t)(b * Nc + n0 + lo) * Cc + h * HDc;
  v16bf a0, a1;
  {
    bf16x8 q00 = *(const bf16x8*)(qrow + hi * 16);
    bf16x8 q01 = *(const bf16x8*)(qrow + 8 + hi * 16);
    bf16x8 q10 = *(const bf16x8*)(qrow + 32 + hi * 16);
    bf16x8 q11 = *(const bf16x8*)(qrow + 40 + hi * 16);
#pragma unroll
    for (int j = 0; j < 8; ++j) {
      a0[j] = q00[j]; a0[j + 8] = q01[j];
      a1[j] = q10[j]; a1[j + 8] = q11[j];
    }
  }

  // ---- Phase 1: logits = scale * Q K^T ------------------------------------
#pragma unroll 1
  for (int it = 0; it < Mc / 128; ++it) {
    const int m0 = it * 128 + wave * 16;
    const __bf16* krow = k_ws + (size_t)(b * Mc + m0 + lo) * Cc + h * HDc;
    // Cover HBM latency for the next iteration's K fragment.
    if (it + 1 < Mc / 128)
      __builtin_prefetch(krow + (size_t)128 * Cc, 0, 0);
    v16bf b0 = *(const v16bf*)(krow + hi * 16);
    v16bf b1 = *(const v16bf*)(krow + 32 + hi * 16);
    v8f acc = {};
    acc = wmma_bf16(a0, b0, acc);
    acc = wmma_bf16(a1, b1, acc);
#pragma unroll
    for (int r = 0; r < 8; ++r)
      s_logits[(size_t)(r + hi * 8) * Mc + m0 + lo] = acc[r] * kScale;
  }
  __syncthreads();

  // ---- Phase 2: softmax over M, entirely in LDS ---------------------------
  {
    const int row = tid >> 4;      // 16 rows x 16 threads each
    const int j16 = tid & 15;
    float mx = -3.4e38f;
    for (int c = j16; c < Mc; c += 16)
      mx = fmaxf(mx, s_logits[(size_t)row * Mc + c]);
    s_red[tid] = mx;
    __syncthreads();
    if (j16 == 0) {
      float m2 = -3.4e38f;
#pragma unroll
      for (int t = 0; t < 16; ++t) m2 = fmaxf(m2, s_red[row * 16 + t]);
      s_stat[row] = m2;
    }
    __syncthreads();
    const float rmax = s_stat[row];
    float sum = 0.0f;
    for (int c = j16; c < Mc; c += 16) {
      const float e = __expf(s_logits[(size_t)row * Mc + c] - rmax);
      s_logits[(size_t)row * Mc + c] = e;
      sum += e;
    }
    s_red[tid] = sum;
    __syncthreads();
    if (j16 == 0) {
      float s2 = 0.0f;
#pragma unroll
      for (int t = 0; t < 16; ++t) s2 += s_red[row * 16 + t];
      s_inv[row] = 1.0f / s2;
    }
    __syncthreads();
  }

  // ---- Write normalized probs: single coalesced non-temporal stream -------
  {
    float* probs = out + (size_t)Bc * Nc * Cc +
                   ((size_t)(b * Hc + h) * Nc + n0) * Mc;
    for (int idx = tid; idx < 16 * Mc; idx += 256) {
      const int r = idx >> 11;           // Mc == 2048
      const int c = idx & (Mc - 1);
      __builtin_nontemporal_store(s_logits[idx] * s_inv[r],
                                  &probs[(size_t)r * Mc + c]);
    }
  }

  // ---- Phase 3: attn_out = P . V  (V pre-transposed: [B,H,64,M]) ----------
  const int dtile = wave & 3;            // 4 x 16 output columns (HD=64)
  const int half  = wave >> 2;           // split M reduction across 2 waves
  const __bf16* vbase =
      vt_ws + ((size_t)((b * Hc + h) * HDc) + dtile * 16 + lo) * Mc;
  const float pinv = s_inv[lo];          // A row == lane&15; fold 1/sum here
  v8f acc = {};
#pragma unroll 1
  for (int ms = 0; ms < 32; ++ms) {
    const int mb = half * 1024 + ms * 32;
    if (ms + 1 < 32)
      __builtin_prefetch(vbase + mb + 32 + hi * 16, 0, 0);
    const float* prow = s_logits + (size_t)lo * Mc + mb + hi * 16;
    f32x8 p0 = *(const f32x8*)prow;
    f32x8 p1 = *(const f32x8*)(prow + 8);
    v16bf a;
#pragma unroll
    for (int j = 0; j < 8; ++j) {
      a[j]     = (__bf16)(p0[j] * pinv);
      a[j + 8] = (__bf16)(p1[j] * pinv);
    }
    v16bf bb = *(const v16bf*)(vbase + mb + hi * 16);
    acc = wmma_bf16(a, bb, acc);
  }
  if (half == 1) {
#pragma unroll
    for (int r = 0; r < 8; ++r)
      s_pv[dtile * 256 + (r + hi * 8) * 16 + lo] = acc[r];
  }
  __syncthreads();
  if (half == 0) {
#pragma unroll
    for (int r = 0; r < 8; ++r) {
      acc[r] += s_pv[dtile * 256 + (r + hi * 8) * 16 + lo];
      const int rr = r + hi * 8;
      attn_ws[(size_t)(b * Nc + n0 + rr) * Cc + h * HDc + dtile * 16 + lo] =
          (__bf16)acc[r];
    }
  }
}

// ---------------------------------------------------------------------------
extern "C" void kernel_launch(void* const* d_in, const int* in_sizes, int n_in,
                              void* d_out, int out_size, void* d_ws,
                              size_t ws_size, hipStream_t stream) {
  const float* query = (const float*)d_in[0];
  const float* key_  = (const float*)d_in[1];
  const float* value = (const float*)d_in[2];
  const float* Wq = (const float*)d_in[3];
  const float* bq = (const float*)d_in[4];
  const float* Wk = (const float*)d_in[5];
  const float* bk = (const float*)d_in[6];
  const float* Wv = (const float*)d_in[7];
  const float* bv = (const float*)d_in[8];
  const float* Wo = (const float*)d_in[9];
  const float* bo = (const float*)d_in[10];
  float* out = (float*)d_out;

  // Workspace: 4 bf16 intermediates of B*N*C = 4M elems = 8 MB each.
  char* ws = (char*)d_ws;
  const size_t kBuf = (size_t)Bc * Nc * Cc * sizeof(__bf16);  // 8 MB
  __bf16* q_ws    = (__bf16*)(ws);
  __bf16* k_ws    = (__bf16*)(ws + kBuf);
  __bf16* vt_ws   = (__bf16*)(ws + 2 * kBuf);
  __bf16* attn_ws = (__bf16*)(ws + 3 * kBuf);

  // Projection grid: (4096/16)*(1024/16) = 16384 tiles, 8 tiles (waves)/block.
  const int proj_blocks = (Bc * Nc / 16) * (Cc / 16) / 8;  // 2048

  proj_kernel<0><<<proj_blocks, 256, 0, stream>>>(query, nullptr, Wq, bq,
                                                  q_ws, nullptr);
  proj_kernel<0><<<proj_blocks, 256, 0, stream>>>(key_, nullptr, Wk, bk,
                                                  k_ws, nullptr);
  proj_kernel<1><<<proj_blocks, 256, 0, stream>>>(value, nullptr, Wv, bv,
                                                  vt_ws, nullptr);

  // Attention: one block per (b, h, 16-row tile) = 2*16*128 = 4096 blocks.
  const int attn_blocks = Bc * Hc * (Nc / 16);
  const size_t lds_bytes =
      (size_t)(16 * Mc + 256 + 16 + 16 + 4 * 256) * sizeof(float);  // ~133 KB
  attention_kernel<<<attn_blocks, 256, lds_bytes, stream>>>(q_ws, k_ws, vt_ws,
                                                            attn_ws, out);

  // Final projection: f32 output region of d_out.
  proj_kernel<2><<<proj_blocks, 256, 0, stream>>>(nullptr, attn_ws, Wo, bo,
                                                  nullptr, out);
}
